// DynamicDecoder_27324581937234
// MI455X (gfx1250) — matrix operand order repaired and checked by reading, not compiled, not run on hardware
//
#include <hip/hip_runtime.h>
#include <math.h>

// ---------------- problem constants (match reference) ----------------
#define HID   256
#define POOL  8
#define NSTEP 4
#define BB    32
#define MM    400
#define DIN   1024          // 4*HID = u_se width
#define NROW  (BB * MM)     // 12800 scored rows
#define W1K   768           // 3*HID
#define W2K   256           // HID
#define W3K   512           // 2*HID
#define NOUT  2048          // HID*POOL
#define NEGBIG (-3.4028234663852886e38f)

// ---------------- CDNA5 WMMA types ----------------
typedef __attribute__((ext_vector_type(16))) __bf16 v16bf;
typedef __attribute__((ext_vector_type(8)))  float  v8f;

union Frag { v16bf v; unsigned int u[8]; };

// fp32 -> bf16 (round to nearest even)
__device__ __forceinline__ unsigned short f2bf(float f) {
  unsigned int x = __float_as_uint(f);
  x += 0x7fffu + ((x >> 16) & 1u);
  return (unsigned short)(x >> 16);
}

__device__ __forceinline__ float sigf(float x) { return 1.0f / (1.0f + __expf(-x)); }

// ---- DPP16 cross-lane max (pure VALU) ----
// 0xB1 = quad_perm[1,0,3,2] (xor1), 0x4E = quad_perm[2,3,0,1] (xor2),
// 0x141 = row_half_mirror (combine 4-groups in 8), 0x140 = row_mirror (combine 8-groups in 16)
template <int CTRL>
__device__ __forceinline__ float maxdpp(float x) {
  int xi = __float_as_int(x);
  int yi = __builtin_amdgcn_update_dpp(xi, xi, CTRL, 0xF, 0xF, true);
  return fmaxf(x, __int_as_float(yi));
}
__device__ __forceinline__ float max8(float v) {
  v = maxdpp<0xB1>(v);
  v = maxdpp<0x4E>(v);
  v = maxdpp<0x141>(v);
  return v;
}
__device__ __forceinline__ float max16(float v) {
  v = max8(v);
  v = maxdpp<0x140>(v);
  return v;
}

// async 16-byte global -> LDS copy (per enabled lane), ASYNCcnt-tracked
__device__ __forceinline__ void async_g2l_b128(unsigned lds_off, const void* gptr) {
  unsigned long long ga = (unsigned long long)(size_t)gptr;
  asm volatile("global_load_async_to_lds_b128 %0, %1, off"
               :: "v"(lds_off), "v"(ga) : "memory");
}
__device__ __forceinline__ void wait_async0() {
  asm volatile("s_wait_asynccnt 0x0" ::: "memory");
}

// A fragment: 16x32 bf16, row-major source (LDS), stride ld (elements, even).
// ISA 7.12.2: lane L: row = L&15, h = L>>4; dword v<4 -> K = 2v+8h ; v>=4 -> K = 16+2(v-4)+8h
__device__ __forceinline__ v16bf load_a_lds(const unsigned short* base, int ld, int kbase, int lane) {
  const int row = lane & 15, h = lane >> 4;
  Frag f;
#pragma unroll
  for (int v = 0; v < 8; ++v) {
    const int k0 = (v < 4) ? (2 * v + 8 * h) : (16 + 2 * (v - 4) + 8 * h);
    f.u[v] = *(const unsigned int*)(base + row * ld + kbase + k0);
  }
  return f.v;
}

// B fragment: 32x16 bf16 from weight matrix stored (out, K) row-major, stride ldk (even).
// lane L: col N = L&15, h = L>>4; dword v<4 -> K = 2v+8h ; v>=4 -> K = 16+2(v-4)+8h
__device__ __forceinline__ v16bf load_b_glb(const unsigned short* w, int ldk, int col0, int kbase, int lane) {
  const int n = lane & 15, h = lane >> 4;
  Frag f;
#pragma unroll
  for (int v = 0; v < 8; ++v) {
    const int k0 = (v < 4) ? (2 * v + 8 * h) : (16 + 2 * (v - 4) + 8 * h);
    f.u[v] = *(const unsigned int*)(w + (size_t)(col0 + n) * ldk + kbase + k0);
  }
  return f.v;
}

// ---------------- small utility kernels ----------------
__global__ void k_cvt_bf16(const float* __restrict__ in, unsigned short* __restrict__ out, int n) {
  int i = blockIdx.x * blockDim.x + threadIdx.x;
  if (i < n) out[i] = f2bf(in[i]);
}

// init: s_idx = 0, e_idx = sum(doc_pad_mask,1)-1, h = c = 0
__global__ void k_init(const int* __restrict__ pad, int* s_idx, int* e_idx,
                       float* h, float* c) {
  int b = blockIdx.x;
  int t = threadIdx.x;
  h[b * HID + t] = 0.0f;
  c[b * HID + t] = 0.0f;
  if (t == 0) {
    int s = 0;
    for (int m = 0; m < MM; ++m) s += pad[b * MM + m];
    e_idx[b] = s - 1;
    s_idx[b] = 0;
  }
}

// gather u_se = [U[b,s_idx[b]], U[b,e_idx[b]]] (fp32, B x 1024)
__global__ void k_gather(const float* __restrict__ U, const int* __restrict__ s_idx,
                         const int* __restrict__ e_idx, float* __restrict__ u_se) {
  int b = blockIdx.x;
  const float* us = U + ((size_t)b * MM + s_idx[b]) * (2 * HID);
  const float* ue = U + ((size_t)b * MM + e_idx[b]) * (2 * HID);
  for (int k = threadIdx.x; k < 2 * HID; k += blockDim.x) {
    u_se[b * DIN + k]           = us[k];
    u_se[b * DIN + 2 * HID + k] = ue[k];
  }
}

// LSTM gate pre-activations: g[b,j] = u_se[b,:] . wih[j,:] + h[b,:] . whh[j,:] + bih[j] + bhh[j]
__global__ void k_lstm_gemm(const float* __restrict__ u_se, const float* __restrict__ h,
                            const float* __restrict__ wih, const float* __restrict__ whh,
                            const float* __restrict__ bih, const float* __restrict__ bhh,
                            float* __restrict__ g) {
  int id = blockIdx.x * blockDim.x + threadIdx.x;
  if (id >= BB * (4 * HID)) return;
  int b = id >> 10, j = id & 1023;
  const float* x  = u_se + b * DIN;
  const float* wr = wih + (size_t)j * DIN;
  float s = bih[j] + bhh[j];
#pragma unroll 4
  for (int k = 0; k < DIN; ++k) s = fmaf(x[k], wr[k], s);
  const float* hh = h + b * HID;
  const float* vr = whh + (size_t)j * HID;
#pragma unroll 4
  for (int k = 0; k < HID; ++k) s = fmaf(hh[k], vr[k], s);
  g[id] = s;
}

__global__ void k_lstm_gates(const float* __restrict__ g, float* __restrict__ h, float* __restrict__ c) {
  int id = blockIdx.x * blockDim.x + threadIdx.x;
  if (id >= BB * HID) return;
  int b = id >> 8, j = id & 255;
  const float* gb = g + b * (4 * HID);
  float ig = sigf(gb[j]);
  float fg = sigf(gb[HID + j]);
  float gg = tanhf(gb[2 * HID + j]);
  float og = sigf(gb[3 * HID + j]);
  float cn = fg * c[id] + ig * gg;
  c[id] = cn;
  h[id] = og * tanhf(cn);
}

// r = tanh([c, u_se] @ wd.T)  -> bf16
__global__ void k_r(const float* __restrict__ c, const float* __restrict__ u_se,
                    const float* __restrict__ wd, unsigned short* __restrict__ rbf) {
  int id = blockIdx.x * blockDim.x + threadIdx.x;
  if (id >= BB * HID) return;
  int b = id >> 8, j = id & 255;
  const float* wr = wd + (size_t)j * (5 * HID);
  float s = 0.0f;
#pragma unroll 4
  for (int k = 0; k < HID; ++k) s = fmaf(c[b * HID + k], wr[k], s);
#pragma unroll 4
  for (int k = 0; k < DIN; ++k) s = fmaf(u_se[b * DIN + k], wr[HID + k], s);
  rbf[id] = f2bf(tanhf(s));
}

// ---------------- fused maxout scorer (WMMA) ----------------
// One workgroup = 128 threads = 4 waves; one 16-row tile of the (B*M) rows.
// Stage1: x(16x768) @ W1.T -> 16x2048, pool8 -> m1 (16x256)  [bf16 in sM12[:, 0:256]]
// Stage2: m1 @ W2.T        -> 16x2048, pool8 -> m2 (16x256)  [bf16 in sM12[:, 256:512]]
// Stage3: [m1,m2] @ W3.T   -> 16x2048, +b3, max over 2048 -> raw score per row.
__global__ __launch_bounds__(128)
void k_score(const unsigned short* __restrict__ Ubf, const unsigned short* __restrict__ rbf,
             const unsigned short* __restrict__ W1, const float* __restrict__ b1,
             const unsigned short* __restrict__ W2, const float* __restrict__ b2,
             const unsigned short* __restrict__ W3, const float* __restrict__ b3,
             float* __restrict__ score) {
  __shared__ __align__(16) unsigned short sX[16 * W1K];     // 24 KB : [U_row(512) | r_b(256)]
  __shared__ __align__(16) unsigned short sM12[16 * W3K];   // 16 KB : [m1(256) | m2(256)] bf16
  __shared__ float sRowMax[4][16];

  const int tid  = threadIdx.x;
  const int lane = tid & 31;
  const int wave = tid >> 5;
  const int rowbase = blockIdx.x * 16;

  // ---- async-stage x rows into LDS: 16 rows x 96 chunks of 16B ----
  for (int cid = tid; cid < 16 * 96; cid += 128) {
    const int r = cid / 96, j = cid - r * 96;
    const int grow = rowbase + r;
    const int b = grow / MM, m = grow - b * MM;
    const unsigned short* src = (j < 64)
        ? Ubf + ((size_t)b * MM + m) * (2 * HID) + j * 8
        : rbf + (size_t)b * HID + (j - 64) * 8;
    const unsigned lds_off = (unsigned)(size_t)(&sX[r * W1K + j * 8]);
    async_g2l_b128(lds_off, src);
  }
  wait_async0();
  __syncthreads();

  const int colq = wave * 512;   // each wave owns 512 of the 2048 output columns

  // ---- stage 1 ----
  for (int ct = 0; ct < 32; ++ct) {
    const int col0 = colq + ct * 16;
    if (ct + 1 < 32) __builtin_prefetch(W1 + (size_t)(col0 + 16) * W1K, 0, 1);
    const float bias = b1[col0 + (lane & 15)];   // issued before the WMMA chain
    v8f acc = {};
#pragma unroll 4
    for (int kt = 0; kt < W1K / 32; ++kt) {
      v16bf a = load_a_lds(sX, W1K, kt * 32, lane);
      v16bf bmat = load_b_glb(W1, W1K, col0, kt * 32, lane);
      acc = __builtin_amdgcn_wmma_f32_16x16x32_bf16(false, a, false, bmat, (short)0, acc, false, false);
    }
    const int u   = (col0 >> 3) + ((lane >> 3) & 1);   // pooled unit 0..255
    const int row8 = 8 * (lane >> 4);
#pragma unroll
    for (int d = 0; d < 8; ++d) {
      float v = max8(acc[d] + bias);                   // pool over 8 adjacent columns (DPP)
      sM12[(d + row8) * W3K + u] = f2bf(v);            // all 8 lanes of a group store same value
    }
  }
  __syncthreads();

  // ---- stage 2 (A = m1, K=256): preload all 8 A fragments into registers ----
  {
    v16bf a2[8];
#pragma unroll
    for (int kt = 0; kt < W2K / 32; ++kt) a2[kt] = load_a_lds(sM12, W3K, kt * 32, lane);

    const int u_lo = ((lane >> 3) & 1);
    const int row8 = 8 * (lane >> 4);
    for (int ct = 0; ct < 32; ++ct) {
      const int col0 = colq + ct * 16;
      if (ct + 1 < 32) __builtin_prefetch(W2 + (size_t)(col0 + 16) * W2K, 0, 1);
      const float bias = b2[col0 + (lane & 15)];
      v8f acc = {};
#pragma unroll
      for (int kt = 0; kt < W2K / 32; ++kt) {
        v16bf bmat = load_b_glb(W2, W2K, col0, kt * 32, lane);
        acc = __builtin_amdgcn_wmma_f32_16x16x32_bf16(false, a2[kt], false, bmat, (short)0, acc, false, false);
      }
      const int u = (col0 >> 3) + u_lo;
#pragma unroll
      for (int d = 0; d < 8; ++d) {
        float v = max8(acc[d] + bias);
        sM12[(d + row8) * W3K + HID + u] = f2bf(v);
      }
    }
  }
  __syncthreads();

  // ---- stage 3 (A = [m1,m2], K=512), max over all 2048 outputs ----
  float rowmax[8];
#pragma unroll
  for (int d = 0; d < 8; ++d) rowmax[d] = NEGBIG;

  for (int ct = 0; ct < 32; ++ct) {
    const int col0 = colq + ct * 16;
    if (ct + 1 < 32) __builtin_prefetch(W3 + (size_t)(col0 + 16) * W3K, 0, 1);
    const float bias = b3[col0 + (lane & 15)];
    v8f acc = {};
#pragma unroll 4
    for (int kt = 0; kt < W3K / 32; ++kt) {
      v16bf a = load_a_lds(sM12, W3K, kt * 32, lane);
      v16bf bmat = load_b_glb(W3, W3K, col0, kt * 32, lane);
      acc = __builtin_amdgcn_wmma_f32_16x16x32_bf16(false, a, false, bmat, (short)0, acc, false, false);
    }
#pragma unroll
    for (int d = 0; d < 8; ++d) {
      float v = max16(acc[d] + bias);          // max over the 16 columns of this tile
      rowmax[d] = fmaxf(rowmax[d], v);
    }
  }
  if ((lane & 15) == 0) {
#pragma unroll
    for (int d = 0; d < 8; ++d) sRowMax[wave][(lane >> 4) * 8 + d] = rowmax[d];
  }
  __syncthreads();
  if (tid < 16) {
    float s = fmaxf(fmaxf(sRowMax[0][tid], sRowMax[1][tid]),
                    fmaxf(sRowMax[2][tid], sRowMax[3][tid]));
    score[rowbase + tid] = s;
  }
}

// ---------------- softmax / argmax / CE / mask update (per batch row) ----------------
__global__ void k_select(const float* __restrict__ score, const int* __restrict__ pad,
                         const int* __restrict__ target, int tcol,
                         int* idx_cur, int* mask_cur,
                         int* res, int* masks, float* nll, int t) {
  const int b = blockIdx.x;
  const int tid = threadIdx.x;
  __shared__ float sv[128];
  __shared__ int   si[128];
  __shared__ float ss[128];

  // pass 1: max & first-argmax of z = score + pad
  float best = NEGBIG; int bi = MM;
  for (int m = tid; m < MM; m += 128) {
    float z = pad[b * MM + m] ? score[b * MM + m] : NEGBIG;
    if (z > best) { best = z; bi = m; }
  }
  sv[tid] = best; si[tid] = bi;
  __syncthreads();
  for (int s = 64; s > 0; s >>= 1) {
    if (tid < s) {
      float o = sv[tid + s]; int oi = si[tid + s];
      if (o > sv[tid] || (o == sv[tid] && oi < si[tid])) { sv[tid] = o; si[tid] = oi; }
    }
    __syncthreads();
  }
  const float maxz = sv[0];
  const int   amax = si[0];
  __syncthreads();

  // pass 2: softmax denominator
  float acc = 0.0f;
  for (int m = tid; m < MM; m += 128) {
    float z = pad[b * MM + m] ? score[b * MM + m] : NEGBIG;
    acc += __expf(z - maxz);
  }
  ss[tid] = acc;
  __syncthreads();
  for (int s = 64; s > 0; s >>= 1) { if (tid < s) ss[tid] += ss[tid + s]; __syncthreads(); }
  const float denom = ss[0];
  __syncthreads();

  // pass 3: sum of exp(p)  (reference applies log_softmax on the softmaxed scores)
  acc = 0.0f;
  for (int m = tid; m < MM; m += 128) {
    float z = pad[b * MM + m] ? score[b * MM + m] : NEGBIG;
    float p = __expf(z - maxz) / denom;
    acc += __expf(p);
  }
  ss[tid] = acc;
  __syncthreads();
  for (int s = 64; s > 0; s >>= 1) { if (tid < s) ss[tid] += ss[tid + s]; __syncthreads(); }

  if (tid == 0) {
    const float sumExpP = ss[0];
    const int tg = target[b * 2 + tcol];
    float zt = pad[b * MM + tg] ? score[b * MM + tg] : NEGBIG;
    float pt = __expf(zt - maxz) / denom;
    nll[t * BB + b] = -(pt - __logf(sumExpP));

    int mnew, inew;
    if (t == 0) {
      mnew = 1; inew = amax;
    } else {
      int mold = mask_cur[b];
      int idx  = amax * mold;
      int prev = idx_cur[b] * mold;
      mnew = (idx != prev) ? 1 : 0;
      inew = idx;
    }
    idx_cur[b]        = inew;
    mask_cur[b]       = mnew;
    res[t * BB + b]   = inew;
    masks[t * BB + b] = mnew;
  }
}

// ---------------- final loss + index selection (1 wave) ----------------
__global__ void k_final(const int* __restrict__ res_s, const int* __restrict__ masks_s,
                        const float* __restrict__ nll_s,
                        const int* __restrict__ res_e, const int* __restrict__ masks_e,
                        const float* __restrict__ nll_e, float* __restrict__ out) {
  const int b = threadIdx.x;  // 0..31, single wave32
  float loss_b = 0.0f;
  int sumMs = 0, sumMe = 0;
#pragma unroll
  for (int t = 0; t < NSTEP; ++t) {
    float vs = nll_s[t * BB + b];
    float ve = nll_e[t * BB + b];
#pragma unroll
    for (int o = 16; o > 0; o >>= 1) { vs += __shfl_xor(vs, o, 32); ve += __shfl_xor(ve, o, 32); }
    const float mean_s = vs * (1.0f / BB);
    const float mean_e = ve * (1.0f / BB);
    loss_b += mean_s * (float)masks_s[t * BB + b] + mean_e * (float)masks_e[t * BB + b];
    sumMs += masks_s[t * BB + b];
    sumMe += masks_e[t * BB + b];
  }
  const int idx_s = res_s[(sumMs - 1) * BB + b];
  const int idx_e = res_e[(sumMe - 1) * BB + b];

  float tot = loss_b;
#pragma unroll
  for (int o = 16; o > 0; o >>= 1) tot += __shfl_xor(tot, o, 32);
  if (b == 0) out[0] = tot * (1.0f / BB) * (1.0f / NSTEP);
  out[1 + b]      = (float)idx_s;
  out[1 + BB + b] = (float)idx_e;
}

// ---------------- host-side orchestration ----------------
static inline size_t alignUp(size_t x) { return (x + 255) & ~(size_t)255; }

extern "C" void kernel_launch(void* const* d_in, const int* in_sizes, int n_in,
                              void* d_out, int out_size, void* d_ws, size_t ws_size,
                              hipStream_t stream) {
  (void)in_sizes; (void)n_in; (void)out_size; (void)ws_size;
  // setup_inputs() dict order:
  const float* U      = (const float*)d_in[0];
  const int*   pad    = (const int*)  d_in[1];
  const int*   target = (const int*)  d_in[2];
  const float* wih    = (const float*)d_in[3];
  const float* whh    = (const float*)d_in[4];
  const float* bih    = (const float*)d_in[5];
  const float* bhh    = (const float*)d_in[6];
  const float* wd_s   = (const float*)d_in[7];
  const float* w1_s   = (const float*)d_in[8];
  const float* b1_s   = (const float*)d_in[9];
  const float* w2_s   = (const float*)d_in[10];
  const float* b2_s   = (const float*)d_in[11];
  const float* w3_s   = (const float*)d_in[12];
  const float* b3_s   = (const float*)d_in[13];
  const float* wd_e   = (const float*)d_in[14];
  const float* w1_e   = (const float*)d_in[15];
  const float* b1_e   = (const float*)d_in[16];
  const float* w2_e   = (const float*)d_in[17];
  const float* b2_e   = (const float*)d_in[18];
  const float* w3_e   = (const float*)d_in[19];
  const float* b3_e   = (const float*)d_in[20];

  // workspace carve-up (~27 MB)
  char* ws = (char*)d_ws;
  size_t off = 0;
  auto take = [&](size_t bytes) { char* p = ws + off; off = alignUp(off + bytes); return p; };

  unsigned short* Ubf  = (unsigned short*)take((size_t)NROW * 2 * HID * 2);
  unsigned short* W1s  = (unsigned short*)take((size_t)NOUT * W1K * 2);
  unsigned short* W2s  = (unsigned short*)take((size_t)NOUT * W2K * 2);
  unsigned short* W3s  = (unsigned short*)take((size_t)NOUT * W3K * 2);
  unsigned short* W1e  = (unsigned short*)take((size_t)NOUT * W1K * 2);
  unsigned short* W2e  = (unsigned short*)take((size_t)NOUT * W2K * 2);
  unsigned short* W3e  = (unsigned short*)take((size_t)NOUT * W3K * 2);
  float* u_se   = (float*)take((size_t)BB * DIN * 4);
  float* g      = (float*)take((size_t)BB * 4 * HID * 4);
  float* h      = (float*)take((size_t)BB * HID * 4);
  float* c      = (float*)take((size_t)BB * HID * 4);
  unsigned short* rbf = (unsigned short*)take((size_t)BB * HID * 2);
  float* score  = (float*)take((size_t)NROW * 4);
  int*   s_idx  = (int*)take(BB * 4);
  int*   e_idx  = (int*)take(BB * 4);
  int*   mask_s = (int*)take(BB * 4);
  int*   mask_e = (int*)take(BB * 4);
  int*   res_s  = (int*)take(NSTEP * BB * 4);
  int*   res_e  = (int*)take(NSTEP * BB * 4);
  int*   msks_s = (int*)take(NSTEP * BB * 4);
  int*   msks_e = (int*)take(NSTEP * BB * 4);
  float* nll_s  = (float*)take(NSTEP * BB * 4);
  float* nll_e  = (float*)take(NSTEP * BB * 4);

  // one-time (per launch) bf16 conversion of U and the scorer weights
  {
    int n = NROW * 2 * HID;
    k_cvt_bf16<<<(n + 255) / 256, 256, 0, stream>>>(U, Ubf, n);
    n = NOUT * W1K;
    k_cvt_bf16<<<(n + 255) / 256, 256, 0, stream>>>(w1_s, W1s, n);
    k_cvt_bf16<<<(n + 255) / 256, 256, 0, stream>>>(w1_e, W1e, n);
    n = NOUT * W2K;
    k_cvt_bf16<<<(n + 255) / 256, 256, 0, stream>>>(w2_s, W2s, n);
    k_cvt_bf16<<<(n + 255) / 256, 256, 0, stream>>>(w2_e, W2e, n);
    n = NOUT * W3K;
    k_cvt_bf16<<<(n + 255) / 256, 256, 0, stream>>>(w3_s, W3s, n);
    k_cvt_bf16<<<(n + 255) / 256, 256, 0, stream>>>(w3_e, W3e, n);
  }

  k_init<<<BB, HID, 0, stream>>>(pad, s_idx, e_idx, h, c);

  for (int t = 0; t < NSTEP; ++t) {
    // u_se = [U[b,s_idx], U[b,e_idx]]
    k_gather<<<BB, 256, 0, stream>>>(U, s_idx, e_idx, u_se);
    // LSTM step (h_i = c)
    k_lstm_gemm<<<(BB * 4 * HID + 255) / 256, 256, 0, stream>>>(u_se, h, wih, whh, bih, bhh, g);
    k_lstm_gates<<<(BB * HID + 255) / 256, 256, 0, stream>>>(g, h, c);

    // ---- start scorer ----
    k_r<<<(BB * HID + 255) / 256, 256, 0, stream>>>(c, u_se, wd_s, rbf);
    k_score<<<NROW / 16, 128, 0, stream>>>(Ubf, rbf, W1s, b1_s, W2s, b2_s, W3s, b3_s, score);
    k_select<<<BB, 128, 0, stream>>>(score, pad, target, 0, s_idx, mask_s, res_s, msks_s, nll_s, t);

    // ---- end scorer (u_se2 uses updated s_idx) ----
    k_gather<<<BB, 256, 0, stream>>>(U, s_idx, e_idx, u_se);
    k_r<<<(BB * HID + 255) / 256, 256, 0, stream>>>(c, u_se, wd_e, rbf);
    k_score<<<NROW / 16, 128, 0, stream>>>(Ubf, rbf, W1e, b1_e, W2e, b2_e, W3e, b3_e, score);
    k_select<<<BB, 128, 0, stream>>>(score, pad, target, 1, e_idx, mask_e, res_e, msks_e, nll_e, t);
  }

  k_final<<<1, BB, 0, stream>>>(res_s, msks_s, nll_s, res_e, msks_e, nll_e, (float*)d_out);
}
